// Transformer_77043123355705
// MI455X (gfx1250) — compile-verified, matrix operand order checked by reading, not run on hardware
//
#include <hip/hip_runtime.h>
#include <stdint.h>

#define S_LEN 4096
#define HIDN  768
#define NH    12
#define DH    64
#define WB    128     // band half-width
#define WIN   320     // key window per 64-query chunk = 64 + 2*WB
#define WSTR  328     // padded LDS stride (ushort elems), multiple of 8

typedef __attribute__((ext_vector_type(8)))  float  v8f;
typedef __attribute__((ext_vector_type(8)))  __bf16 v8bf;
typedef __attribute__((ext_vector_type(16))) __bf16 v16bf;

__device__ __forceinline__ unsigned short f2bf(float f) {
  unsigned u = __float_as_uint(f);
  u += 0x7FFFu + ((u >> 16) & 1u);      // round-to-nearest-even
  return (unsigned short)(u >> 16);
}

// Load one 16x32 bf16 WMMA operand fragment (A layout; B uses the same
// pattern with lane%16 = N column index). Row-major source, `stride` in
// elements. lane<16 -> K = {k0..k0+7, k0+16..k0+23}; lane>=16 -> +8.
__device__ __forceinline__ v16bf load_frag(const unsigned short* base, int row,
                                           int stride, int k0, int lane) {
  const unsigned short* p = base + (size_t)row * stride + k0 + ((lane >> 4) << 3);
  v8bf lo = *reinterpret_cast<const v8bf*>(p);
  v8bf hi = *reinterpret_cast<const v8bf*>(p + 16);
  return __builtin_shufflevector(lo, hi, 0,1,2,3,4,5,6,7,8,9,10,11,12,13,14,15);
}

__device__ __forceinline__ v8f wmma_bf16(v16bf a, v16bf b, v8f c) {
  return __builtin_amdgcn_wmma_f32_16x16x32_bf16(false, a, false, b, (short)0, c,
                                                 false, false);
}

// ---------------- conversion kernels ----------------
__global__ void cvt_f32_bf16(const float* __restrict__ src,
                             unsigned short* __restrict__ dst, int n) {
  int i = (blockIdx.x * blockDim.x + threadIdx.x) * 4;
  if (i + 3 < n) {
    float4 v = *reinterpret_cast<const float4*>(src + i);
    dst[i + 0] = f2bf(v.x);
    dst[i + 1] = f2bf(v.y);
    dst[i + 2] = f2bf(v.z);
    dst[i + 3] = f2bf(v.w);
  }
}

// dst[o][i] = src[i][o]  (768x768)
__global__ void cvt_transpose768(const float* __restrict__ src,
                                 unsigned short* __restrict__ dst) {
  int e = blockIdx.x * blockDim.x + threadIdx.x;
  int i = e / HIDN, o = e % HIDN;
  dst[o * HIDN + i] = f2bf(src[e]);
}

// ---------------- QKV projection GEMM ----------------
// out[row][col] = bf16( (sum_k xb[row][k]*WT[col][k] + bias[col]) * scale )
// grid: (M/128, 768/128); block 256 (8 waves); wave = 16 rows x 128 cols.
__global__ __launch_bounds__(256) void gemm_qkv(
    const unsigned short* __restrict__ xb, const unsigned short* __restrict__ wt,
    const float* __restrict__ bias, unsigned short* __restrict__ dst, float scale) {
  int lane = threadIdx.x & 31, wv = threadIdx.x >> 5;
  int mbase = blockIdx.x * 128 + wv * 16;
  int nbase = blockIdx.y * 128;
  v8f acc[8];
#pragma unroll
  for (int t = 0; t < 8; ++t) acc[t] = (v8f){0,0,0,0,0,0,0,0};
  for (int k0 = 0; k0 < HIDN; k0 += 32) {
    v16bf a = load_frag(xb, mbase + (lane & 15), HIDN, k0, lane);
#pragma unroll
    for (int t = 0; t < 8; ++t) {
      v16bf b = load_frag(wt, nbase + t * 16 + (lane & 15), HIDN, k0, lane);
      acc[t] = wmma_bf16(a, b, acc[t]);
    }
  }
#pragma unroll
  for (int t = 0; t < 8; ++t) {
#pragma unroll
    for (int j = 0; j < 8; ++j) {
      int col = nbase + t * 16 + (lane & 15);
      int row = mbase + ((lane >> 4) << 3) + j;
      dst[(size_t)row * HIDN + col] = f2bf((acc[t][j] + bias[col]) * scale);
    }
  }
}

// ---------------- banded attention ----------------
// grid: (S/64, NH, B); block 256 = 8 waves.
// wave (mg=wv>>1, ch=wv&1): scores for rows [16*mg,16*mg+16) x cols [160*ch,+160).
__global__ __launch_bounds__(256) void banded_attn(
    const unsigned short* __restrict__ qb, const unsigned short* __restrict__ kb,
    const unsigned short* __restrict__ vb, unsigned short* __restrict__ ctx) {
  __shared__ unsigned short Psm[64 * WSTR];
  __shared__ unsigned short VTsm[64 * WSTR];
  __shared__ float redmax[64][2];
  __shared__ float redsum[64][2];

  int lane = threadIdx.x & 31, wv = threadIdx.x >> 5;
  int c = blockIdx.x, h = blockIdx.y, b = blockIdx.z;
  int q0 = c * 64;
  int jbase = q0 - WB;

  const unsigned short* qbh = qb + (size_t)b * S_LEN * HIDN + h * DH;
  const unsigned short* kbh = kb + (size_t)b * S_LEN * HIDN + h * DH;
  const unsigned short* vbh = vb + (size_t)b * S_LEN * HIDN + h * DH;

  // stage V^T : VTsm[d][j] = V[jbase+j][d]
  for (int e = threadIdx.x; e < 64 * WIN; e += 256) {
    int j = e >> 6, d = e & 63;
    int ja = jbase + j;
    int jr = ja < 0 ? 0 : (ja >= S_LEN ? S_LEN - 1 : ja);
    VTsm[d * WSTR + j] = vbh[(size_t)jr * HIDN + d];
  }

  int mg = wv >> 1;
  int ch = wv & 1;
  int colb = ch * 160;

  v16bf aq0 = load_frag(qbh, q0 + mg * 16 + (lane & 15), HIDN, 0, lane);
  v16bf aq1 = load_frag(qbh, q0 + mg * 16 + (lane & 15), HIDN, 32, lane);

  v8f acc[10];
#pragma unroll
  for (int t = 0; t < 10; ++t) acc[t] = (v8f){0,0,0,0,0,0,0,0};

#pragma unroll
  for (int t = 0; t < 10; ++t) {
    int jn = colb + t * 16 + (lane & 15);
    int ja = jbase + jn;
    int jr = ja < 0 ? 0 : (ja >= S_LEN ? S_LEN - 1 : ja);
    v16bf b0 = load_frag(kbh, jr, HIDN, 0, lane);
    v16bf b1 = load_frag(kbh, jr, HIDN, 32, lane);
    acc[t] = wmma_bf16(aq0, b0, acc[t]);
    acc[t] = wmma_bf16(aq1, b1, acc[t]);
  }

  // band + sequence-bounds mask
#pragma unroll
  for (int t = 0; t < 10; ++t) {
    int jn = colb + t * 16 + (lane & 15);
    int ja = jbase + jn;
#pragma unroll
    for (int j = 0; j < 8; ++j) {
      int x = mg * 16 + ((lane >> 4) << 3) + j;   // local query row
      bool ok = (jn >= x) && (jn <= x + 2 * WB) && (ja >= 0) && (ja < S_LEN);
      if (!ok) acc[t][j] = -1e9f;
    }
  }

  // row max (16-lane halves hold rows M and M+8)
  float pm[8];
#pragma unroll
  for (int j = 0; j < 8; ++j) {
    float m = acc[0][j];
#pragma unroll
    for (int t = 1; t < 10; ++t) m = fmaxf(m, acc[t][j]);
#pragma unroll
    for (int s = 1; s < 16; s <<= 1) m = fmaxf(m, __shfl_xor(m, s, 32));
    pm[j] = m;
  }
  if ((lane & 15) == 0) {
#pragma unroll
    for (int j = 0; j < 8; ++j)
      redmax[mg * 16 + ((lane >> 4) << 3) + j][ch] = pm[j];
  }
  __syncthreads();

  float gm[8], ps[8];
#pragma unroll
  for (int j = 0; j < 8; ++j) {
    int r = mg * 16 + ((lane >> 4) << 3) + j;
    gm[j] = fmaxf(redmax[r][0], redmax[r][1]);
  }
#pragma unroll
  for (int j = 0; j < 8; ++j) {
    float s = 0.f;
#pragma unroll
    for (int t = 0; t < 10; ++t) { acc[t][j] = __expf(acc[t][j] - gm[j]); s += acc[t][j]; }
#pragma unroll
    for (int sh = 1; sh < 16; sh <<= 1) s += __shfl_xor(s, sh, 32);
    ps[j] = s;
  }
  if ((lane & 15) == 0) {
#pragma unroll
    for (int j = 0; j < 8; ++j)
      redsum[mg * 16 + ((lane >> 4) << 3) + j][ch] = ps[j];
  }
  __syncthreads();

#pragma unroll
  for (int j = 0; j < 8; ++j) {
    int r = mg * 16 + ((lane >> 4) << 3) + j;
    float inv = 1.f / (redsum[r][0] + redsum[r][1]);
#pragma unroll
    for (int t = 0; t < 10; ++t)
      Psm[r * WSTR + colb + t * 16 + (lane & 15)] = f2bf(acc[t][j] * inv);
  }
  __syncthreads();

  // P @ V : wave -> (mt = wv>>1) rows, d-tiles {2*(wv&1), 2*(wv&1)+1}
  int mt = wv >> 1;
  int dp = wv & 1;
  v8f c0 = (v8f){0,0,0,0,0,0,0,0}, c1 = (v8f){0,0,0,0,0,0,0,0};
  for (int ks = 0; ks < WIN; ks += 32) {
    v16bf a  = load_frag(Psm,  mt * 16 + (lane & 15),            WSTR, ks, lane);
    v16bf b0 = load_frag(VTsm, (dp * 2 + 0) * 16 + (lane & 15),  WSTR, ks, lane);
    v16bf b1 = load_frag(VTsm, (dp * 2 + 1) * 16 + (lane & 15),  WSTR, ks, lane);
    c0 = wmma_bf16(a, b0, c0);
    c1 = wmma_bf16(a, b1, c1);
  }
#pragma unroll
  for (int j = 0; j < 8; ++j) {
    size_t row = (size_t)b * S_LEN + q0 + mt * 16 + ((lane >> 4) << 3) + j;
    ctx[row * HIDN + h * DH + (dp * 2 + 0) * 16 + (lane & 15)] = f2bf(c0[j]);
    ctx[row * HIDN + h * DH + (dp * 2 + 1) * 16 + (lane & 15)] = f2bf(c1[j]);
  }
}

// ---------------- output projection + residual + LayerNorm ----------------
// grid: (B*S/16); block 256. wave wv -> cols [96*wv, 96*wv+96).
__global__ __launch_bounds__(256) void proj_ln(
    const unsigned short* __restrict__ ctx, const unsigned short* __restrict__ wdt,
    const float* __restrict__ bd, const float* __restrict__ x,
    const float* __restrict__ gamma, const float* __restrict__ beta,
    float* __restrict__ out) {
  __shared__ float ysm[16 * HIDN];
  int lane = threadIdx.x & 31, wv = threadIdx.x >> 5;
  int m0 = blockIdx.x * 16;
  v8f acc[6];
#pragma unroll
  for (int t = 0; t < 6; ++t) acc[t] = (v8f){0,0,0,0,0,0,0,0};
  for (int k0 = 0; k0 < HIDN; k0 += 32) {
    v16bf a = load_frag(ctx, m0 + (lane & 15), HIDN, k0, lane);
#pragma unroll
    for (int t = 0; t < 6; ++t) {
      v16bf b = load_frag(wdt, wv * 96 + t * 16 + (lane & 15), HIDN, k0, lane);
      acc[t] = wmma_bf16(a, b, acc[t]);
    }
  }
#pragma unroll
  for (int t = 0; t < 6; ++t) {
#pragma unroll
    for (int j = 0; j < 8; ++j) {
      int col = wv * 96 + t * 16 + (lane & 15);
      int rl = ((lane >> 4) << 3) + j;
      ysm[rl * HIDN + col] = acc[t][j] + bd[col] + x[(size_t)(m0 + rl) * HIDN + col];
    }
  }
  __syncthreads();
#pragma unroll
  for (int rr = wv * 2; rr < wv * 2 + 2; ++rr) {
    float s = 0.f;
#pragma unroll
    for (int i = 0; i < 24; ++i) s += ysm[rr * HIDN + lane + 32 * i];
#pragma unroll
    for (int sh = 1; sh < 32; sh <<= 1) s += __shfl_xor(s, sh, 32);
    float mu = s * (1.f / HIDN);
    float vs = 0.f;
#pragma unroll
    for (int i = 0; i < 24; ++i) {
      float d = ysm[rr * HIDN + lane + 32 * i] - mu;
      vs += d * d;
    }
#pragma unroll
    for (int sh = 1; sh < 32; sh <<= 1) vs += __shfl_xor(vs, sh, 32);
    float rs = rsqrtf(vs * (1.f / HIDN) + 1e-12f);
#pragma unroll
    for (int i = 0; i < 24; ++i) {
      int col = lane + 32 * i;
      out[(size_t)(m0 + rr) * HIDN + col] =
          (ysm[rr * HIDN + col] - mu) * rs * gamma[col] + beta[col];
    }
  }
}

// ---------------- launch ----------------
extern "C" void kernel_launch(void* const* d_in, const int* in_sizes, int n_in,
                              void* d_out, int out_size, void* d_ws, size_t ws_size,
                              hipStream_t stream) {
  (void)in_sizes; (void)n_in; (void)out_size; (void)ws_size;
  const float* x     = (const float*)d_in[0];
  const float* Wq    = (const float*)d_in[1];
  const float* bq    = (const float*)d_in[2];
  const float* Wk    = (const float*)d_in[3];
  const float* bk    = (const float*)d_in[4];
  const float* Wv    = (const float*)d_in[5];
  const float* bv    = (const float*)d_in[6];
  const float* Wd    = (const float*)d_in[7];
  const float* bd    = (const float*)d_in[8];
  const float* gamma = (const float*)d_in[9];
  const float* beta  = (const float*)d_in[10];
  float* out = (float*)d_out;

  char* ws = (char*)d_ws;
  const size_t ROWS = 2 * (size_t)S_LEN;                 // 8192
  const size_t XBYT = ROWS * HIDN * 2;                   // 12,582,912
  const size_t WBYT = (size_t)HIDN * HIDN * 2;           // 1,179,648
  unsigned short* xb  = (unsigned short*)(ws);
  unsigned short* wqt = (unsigned short*)(ws + XBYT);
  unsigned short* wkt = (unsigned short*)(ws + XBYT + WBYT);
  unsigned short* wvt = (unsigned short*)(ws + XBYT + 2 * WBYT);
  unsigned short* wdt = (unsigned short*)(ws + XBYT + 3 * WBYT);
  unsigned short* qb  = (unsigned short*)(ws + XBYT + 4 * WBYT);
  unsigned short* kb  = (unsigned short*)(ws + 2 * XBYT + 4 * WBYT);
  unsigned short* vb  = (unsigned short*)(ws + 3 * XBYT + 4 * WBYT);
  unsigned short* ctx = (unsigned short*)(ws + 4 * XBYT + 4 * WBYT);

  int nx = (int)(ROWS * HIDN);
  cvt_f32_bf16<<<nx / 1024, 256, 0, stream>>>(x, xb, nx);
  cvt_transpose768<<<(HIDN * HIDN) / 256, 256, 0, stream>>>(Wq, wqt);
  cvt_transpose768<<<(HIDN * HIDN) / 256, 256, 0, stream>>>(Wk, wkt);
  cvt_transpose768<<<(HIDN * HIDN) / 256, 256, 0, stream>>>(Wv, wvt);
  cvt_transpose768<<<(HIDN * HIDN) / 256, 256, 0, stream>>>(Wd, wdt);

  gemm_qkv<<<dim3(ROWS / 128, HIDN / 128), 256, 0, stream>>>(xb, wqt, bq, qb, 0.125f);
  gemm_qkv<<<dim3(ROWS / 128, HIDN / 128), 256, 0, stream>>>(xb, wkt, bk, kb, 1.0f);
  gemm_qkv<<<dim3(ROWS / 128, HIDN / 128), 256, 0, stream>>>(xb, wvt, bv, vb, 1.0f);

  banded_attn<<<dim3(S_LEN / 64, NH, 2), 256, 0, stream>>>(qb, kb, vb, ctx);

  proj_ln<<<ROWS / 16, 256, 0, stream>>>(ctx, wdt, bd, x, gamma, beta, out);
}
